// BiRNN_CRF_76227079570006
// MI455X (gfx1250) — matrix operand order repaired and checked by reading, not compiled, not run on hardware
//
#include <hip/hip_runtime.h>
#include <hip/hip_bf16.h>

// ---------------------------------------------------------------------------
// BiLSTM(H=256)+CRF(T=9), B=64, S=512, E=768.
// Phase 1: bf16 WMMA GEMM  pre = X @ [w_ih_f;w_ih_b]^T + b      (32768x768x2048)
//          A/B tiles staged by Tensor Data Mover (double-buffered LDS)
// Phase 2: persistent-WG recurrent LSTM, WMMA per step, global step barrier,
//          h staged by TDM
// Phase 3: projection to 9 tags, CRF forward algorithm, scalar reduce
// ---------------------------------------------------------------------------

typedef __attribute__((ext_vector_type(16))) __bf16          v16bf;
typedef __attribute__((ext_vector_type(8)))  float           v8f;
typedef __attribute__((ext_vector_type(8)))  unsigned short  v8us;
typedef __attribute__((ext_vector_type(16))) unsigned short  v16us;

__device__ __forceinline__ unsigned short f2bf(float f) {
    unsigned int u = __float_as_uint(f);
    unsigned int r = u + 0x7fffu + ((u >> 16) & 1u);   // round-to-nearest-even
    return (unsigned short)(r >> 16);
}

__device__ __forceinline__ v16bf frag16(const unsigned short* p0,
                                        const unsigned short* p1) {
    v8us lo = *(const v8us*)p0;
    v8us hi = *(const v8us*)p1;
    v16us w = __builtin_shufflevector(lo, hi,
              0,1,2,3,4,5,6,7,8,9,10,11,12,13,14,15);
    return __builtin_bit_cast(v16bf, w);
}

__device__ __forceinline__ float sigm(float x) { return 1.0f / (1.0f + __expf(-x)); }

// ------------------------- Tensor Data Mover helpers ------------------------
#if __has_builtin(__builtin_amdgcn_tensor_load_to_lds)
#define HAVE_TDM 1
typedef __attribute__((ext_vector_type(4))) unsigned int u32x4;
typedef __attribute__((ext_vector_type(8))) int          i32x8;
typedef __attribute__((ext_vector_type(4))) int          i32x4;

// 2D tile load (bf16 elements): tile_w x tile_h from a row-major tensor
// (row stride = stride_elems), into LDS at lds_byte.  Optional LDS row
// padding: pad_amount+1 DWORDs inserted after every (2<<pad_interval) DWORDs.
__device__ __forceinline__ void tdm_load_2d(const void* gptr, unsigned lds_byte,
                                            unsigned tile_w, unsigned tile_h,
                                            unsigned tensor_w, unsigned tensor_h,
                                            unsigned stride_elems,
                                            unsigned pad_interval, unsigned pad_amount,
                                            bool pad_en) {
    unsigned long long ga = reinterpret_cast<unsigned long long>(gptr);
    u32x4 g0;
    g0[0] = 1u;                                            // count=1, user D#
    g0[1] = lds_byte;                                      // lds_addr
    g0[2] = (unsigned)(ga & 0xffffffffu);                  // global_addr lo
    g0[3] = (unsigned)((ga >> 32) & 0x1ffffffu) | (2u << 30); // addr hi | type=2
    i32x8 g1;
    unsigned w0 = (1u << 16);                              // data_size=1 -> 2 bytes
    if (pad_en) w0 |= (1u << 20) | (pad_interval << 22) | (pad_amount << 25);
    g1[0] = (int)w0;
    g1[1] = (int)((tensor_w & 0xffffu) << 16);                                  // dim0[15:0]
    g1[2] = (int)(((tensor_w >> 16) & 0xffffu) | ((tensor_h & 0xffffu) << 16)); // dim0 hi|dim1 lo
    g1[3] = (int)(((tensor_h >> 16) & 0xffffu) | ((tile_w & 0xffffu) << 16));   // dim1 hi|tile0
    g1[4] = (int)(tile_h & 0xffffu);                        // tile_dim1 (tile_dim2=0)
    g1[5] = (int)stride_elems;                              // dim0_stride[31:0]
    g1[6] = 0;                                              // dim0_stride hi | dim1_stride lo
    g1[7] = 0;
    i32x4 z4 = {0, 0, 0, 0};
    i32x8 z8 = {0, 0, 0, 0, 0, 0, 0, 0};
    __builtin_amdgcn_tensor_load_to_lds(g0, g1, z4, z4, z8, 0);
}

__device__ __forceinline__ unsigned lds_off(const void* p) {
    return (unsigned)reinterpret_cast<unsigned long long>(p);
}
#else
#define HAVE_TDM 0
#endif

// ------------------------------ init ---------------------------------------
__global__ void init_kernel(unsigned int* bar) {
    if (threadIdx.x == 0) bar[0] = 0u;
}

// ------------------------- conversions --------------------------------------
// Wcat[2048][768] = [w_ih_f ; w_ih_b] (bf16);  Whh_{f,b}[1024][256] bf16
__global__ __launch_bounds__(256)
void convert_weights(const float* __restrict__ wihf, const float* __restrict__ wihb,
                     const float* __restrict__ whhf, const float* __restrict__ whhb,
                     unsigned short* __restrict__ Wcat,
                     unsigned short* __restrict__ Whf,
                     unsigned short* __restrict__ Whb) {
    const int NIH = 1024 * 768;
    const int NHH = 1024 * 256;
    int i = blockIdx.x * 256 + threadIdx.x;
    if (i < NIH)             Wcat[i] = f2bf(wihf[i]);
    else if (i < 2 * NIH)    Wcat[i] = f2bf(wihb[i - NIH]);
    else {
        int k = i - 2 * NIH;
        if (k < NHH)         Whf[k] = f2bf(whhf[k]);
        else if (k < 2*NHH)  Whb[k - NHH] = f2bf(whhb[k - NHH]);
    }
}

// X (32768x768 fp32) -> Xbf (bf16), 8 elements/thread
__global__ __launch_bounds__(256)
void convert_x(const float* __restrict__ X, unsigned short* __restrict__ Xbf) {
    size_t i = ((size_t)blockIdx.x * 256 + threadIdx.x) * 8;
    if (i >= (size_t)32768 * 768) return;
    float4 a = *(const float4*)(X + i);
    float4 b = *(const float4*)(X + i + 4);
    v8us o;
    o[0] = f2bf(a.x); o[1] = f2bf(a.y); o[2] = f2bf(a.z); o[3] = f2bf(a.w);
    o[4] = f2bf(b.x); o[5] = f2bf(b.y); o[6] = f2bf(b.z); o[7] = f2bf(b.w);
    *(v8us*)(Xbf + i) = o;
}

// ------------------------- input-projection GEMM ----------------------------
// PRE[m][n] = sum_k Xbf[m][k] * Wcat[n][k] + bias(n),  m<32768, n<2048, k<768
// Block: 128 thr = 4 waves, tile 64x64, BK=32, bf16 WMMA f32-acc.
// TDM stages 64x32 bf16 tiles into padded [64][40] LDS (80B row stride:
// 16B-aligned fragments, conflict-free banks), double-buffered.
__global__ __launch_bounds__(128)
void gemm_pre(const unsigned short* __restrict__ X,
              const unsigned short* __restrict__ W,
              const float* __restrict__ bias_f, const float* __restrict__ bias_b,
              float* __restrict__ PRE) {
    __shared__ __align__(16) unsigned short As[2][64][40];
    __shared__ __align__(16) unsigned short Bs[2][64][40];

    const int tid  = threadIdx.x;
    const int wave = tid >> 5;
    const int lane = tid & 31;
    const int hl   = lane & 15;
    const int hi   = lane >> 4;

    const int mb = blockIdx.x >> 5;       // 512 M tiles
    const int nb = blockIdx.x & 31;       // 32 N tiles
    const int m0 = mb * 64;
    const int n0 = nb * 64;

    v8f acc[4] = {};
    const int cb0  = hi ? 8 : 0;
    const int hi16 = hi << 4;
    const int arow = (wave << 4) + hl;

#if HAVE_TDM
    const unsigned ldsA[2] = { lds_off(&As[0][0][0]), lds_off(&As[1][0][0]) };
    const unsigned ldsB[2] = { lds_off(&Bs[0][0][0]), lds_off(&Bs[1][0][0]) };
    if (wave == 0) {
        // preload k0 = 0 and k0 = 32 (pad: 16 DWORDs -> +4 DWORDs => 40-short rows)
        tdm_load_2d(X + (size_t)m0 * 768,      ldsA[0], 32, 64, 768, 32768, 768, 3, 3, true);
        tdm_load_2d(W + (size_t)n0 * 768,      ldsB[0], 32, 64, 768, 2048,  768, 3, 3, true);
        tdm_load_2d(X + (size_t)m0 * 768 + 32, ldsA[1], 32, 64, 768, 32768, 768, 3, 3, true);
        tdm_load_2d(W + (size_t)n0 * 768 + 32, ldsB[1], 32, 64, 768, 2048,  768, 3, 3, true);
    }
    int buf = 0;
    for (int k0 = 0; k0 < 768; k0 += 32) {
        if (wave == 0) {
            if (k0 + 32 < 768) __builtin_amdgcn_s_wait_tensorcnt(2);
            else               __builtin_amdgcn_s_wait_tensorcnt(0);
        }
        __syncthreads();

        v16bf a = frag16(&As[buf][arow][cb0], &As[buf][arow][cb0 + 16]);
#pragma unroll
        for (int nt = 0; nt < 4; nt++) {
            v16bf b = frag16(&Bs[buf][nt * 16 + hl][cb0],
                             &Bs[buf][nt * 16 + hl][cb0 + 16]);
            acc[nt] = __builtin_amdgcn_wmma_f32_16x16x32_bf16(
                          false, a, false, b, (short)0, acc[nt], false, false);
        }
        __syncthreads();
        if (wave == 0 && k0 + 64 < 768) {
            tdm_load_2d(X + (size_t)m0 * 768 + k0 + 64, ldsA[buf], 32, 64, 768, 32768, 768, 3, 3, true);
            tdm_load_2d(W + (size_t)n0 * 768 + k0 + 64, ldsB[buf], 32, 64, 768, 2048,  768, 3, 3, true);
        }
        buf ^= 1;
    }
#else
    const int srow = tid >> 1;            // staging row 0..63
    const int kseg = (tid & 1) << 4;      // 0 or 16
    const unsigned short* xp = X + (size_t)(m0 + srow) * 768 + kseg;
    const unsigned short* wp = W + (size_t)(n0 + srow) * 768 + kseg;
    for (int k0 = 0; k0 < 768; k0 += 32) {
        if (k0) __syncthreads();
        uint4 a0 = *(const uint4*)(xp + k0);
        uint4 a1 = *(const uint4*)(xp + k0 + 8);
        uint4 w0 = *(const uint4*)(wp + k0);
        uint4 w1 = *(const uint4*)(wp + k0 + 8);
        *(uint4*)&As[0][srow][kseg]     = a0;
        *(uint4*)&As[0][srow][kseg + 8] = a1;
        *(uint4*)&Bs[0][srow][kseg]     = w0;
        *(uint4*)&Bs[0][srow][kseg + 8] = w1;
        __syncthreads();
        v16bf a = frag16(&As[0][arow][cb0], &As[0][arow][cb0 + 16]);
#pragma unroll
        for (int nt = 0; nt < 4; nt++) {
            v16bf b = frag16(&Bs[0][nt * 16 + hl][cb0], &Bs[0][nt * 16 + hl][cb0 + 16]);
            acc[nt] = __builtin_amdgcn_wmma_f32_16x16x32_bf16(
                          false, a, false, b, (short)0, acc[nt], false, false);
        }
    }
#endif

#pragma unroll
    for (int nt = 0; nt < 4; nt++) {
        int n = n0 + nt * 16 + hl;
        float bias = (n < 1024) ? bias_f[n] : bias_b[n - 1024];
#pragma unroll
        for (int r = 0; r < 8; r++) {
            int m = m0 + (wave << 4) + (hi << 3) + r;
            PRE[(size_t)m * 2048 + n] = acc[nt][r] + bias;
        }
    }
}

// ------------------------- recurrent LSTM -----------------------------------
// 16 persistent WGs: dir = blk>>3 (0 fwd, 1 bwd), chunk = blk&7 owns h cols
// [32*chunk, +32). Per step: G(64x128) = PRE + h(64x256,bf16) @ Whh_chunk^T,
// gates -> (h,c); h published bf16 to double-buffered HBUF; global barrier.
__global__ __launch_bounds__(256)
void lstm_kernel(const float* __restrict__ PRE,
                 const unsigned short* __restrict__ Whh_f,
                 const unsigned short* __restrict__ Whh_b,
                 unsigned short* __restrict__ HBUF,   // [2][2][64][256] bf16
                 float* __restrict__ HSEQ,            // [2][64][512][256]
                 unsigned int* __restrict__ bar) {
    const int dir   = blockIdx.x >> 3;
    const int chunk = blockIdx.x & 7;
    const int j0    = chunk * 32;
    const int tid   = threadIdx.x;
    const int wave  = tid >> 5;
    const int lane  = tid & 31;
    const int hl    = lane & 15;
    const int hi    = lane >> 4;
    const unsigned short* Whh = dir ? Whh_b : Whh_f;

    __shared__ __align__(16) float          c_s[64][32];
    __shared__ __align__(16) unsigned short h_s[64][264];
    __shared__ __align__(16) float          g_s[64][128];

    for (int i = tid; i < 64 * 32; i += 256) ((float*)c_s)[i] = 0.0f;

    const int mt   = wave & 3;                 // row tile (batch 16*mt..)
    const int ntb  = (wave >> 2) * 4;          // first of 4 N tiles
    const int arow = mt * 16 + hl;
    const int cb0  = hi ? 8 : 0;

    for (int t = 0; t < 512; t++) {
        const int sidx = dir ? (511 - t) : t;

        // ---- stage previous h (bf16) into LDS ----
        if (t == 0) {
            for (int i = tid; i < 64 * 264; i += 256) ((unsigned short*)h_s)[i] = 0;
        } else {
            const unsigned short* hsrc = HBUF + ((size_t)(dir * 2 + ((t - 1) & 1))) * 64 * 256;
#if HAVE_TDM
            if (wave == 0) {
                // 64x256 bf16 tile; pad 4 DWORDs after every 128 DWORDs -> 264-short rows
                tdm_load_2d(hsrc, lds_off(&h_s[0][0]), 256, 64, 256, 64, 256, 6, 3, true);
                __builtin_amdgcn_s_wait_tensorcnt(0);
            }
#else
            for (int i = tid; i < 2048; i += 256) {           // 2048 x uint4
                uint4 v = ((const uint4*)hsrc)[i];
                int e   = i * 8;
                *(uint4*)&h_s[e >> 8][e & 255] = v;
            }
#endif
        }
        __syncthreads();

        // ---- G = PRE_tile + h @ Whh^T ----
        v8f acc[4];
#pragma unroll
        for (int nn = 0; nn < 4; nn++) {
            int ntile = ntb + nn;
            int gate  = ntile >> 1, sub = ntile & 1;
            int n_g   = gate * 256 + j0 + sub * 16 + hl;
#pragma unroll
            for (int r = 0; r < 8; r++) {
                int m = mt * 16 + hi * 8 + r;
                acc[nn][r] = PRE[((size_t)m * 512 + sidx) * 2048 + dir * 1024 + n_g];
            }
        }
#pragma unroll
        for (int k0 = 0; k0 < 256; k0 += 32) {
            v16bf a = frag16(&h_s[arow][k0 + cb0], &h_s[arow][k0 + cb0 + 16]);
#pragma unroll
            for (int nn = 0; nn < 4; nn++) {
                int ntile = ntb + nn;
                int gate  = ntile >> 1, sub = ntile & 1;
                int n_g   = gate * 256 + j0 + sub * 16 + hl;
                const unsigned short* wrow = Whh + (size_t)n_g * 256 + k0 + cb0;
                v16bf b = frag16(wrow, wrow + 16);
                acc[nn] = __builtin_amdgcn_wmma_f32_16x16x32_bf16(
                              false, a, false, b, (short)0, acc[nn], false, false);
            }
        }
#pragma unroll
        for (int nn = 0; nn < 4; nn++) {
            int ntile = ntb + nn;
#pragma unroll
            for (int r = 0; r < 8; r++)
                g_s[mt * 16 + hi * 8 + r][ntile * 16 + hl] = acc[nn][r];
        }
        __syncthreads();

        // ---- gates, state update, publish h ----
        for (int p = tid; p < 64 * 32; p += 256) {
            int m = p >> 5, j = p & 31;
            float gi = g_s[m][j],      gf = g_s[m][32 + j];
            float gg = g_s[m][64 + j], go = g_s[m][96 + j];
            float c  = sigm(gf) * c_s[m][j] + sigm(gi) * tanhf(gg);
            float h  = sigm(go) * tanhf(c);
            c_s[m][j] = c;
            HSEQ[(((size_t)dir * 64 + m) * 512 + sidx) * 256 + j0 + j] = h;
            HBUF[((size_t)(dir * 2 + (t & 1)) * 64 + m) * 256 + j0 + j] = f2bf(h);
        }
        // prefetch next step's PRE rows while waiting at the barrier
        if (t < 511 && tid < 64) {
            int nsidx = dir ? (510 - t) : (t + 1);
            __builtin_prefetch(&PRE[((size_t)tid * 512 + nsidx) * 2048 + dir * 1024 + j0], 0, 0);
        }
        __syncthreads();

        // ---- grid-wide step barrier (monotonic counter in ws) ----
        if (tid == 0) {
            __threadfence();
            atomicAdd(bar, 1u);
            unsigned target = 16u * (unsigned)(t + 1);
            while (*(volatile unsigned int*)bar < target)
                __builtin_amdgcn_s_sleep(1);
            __threadfence();
        }
        __syncthreads();
    }
}

// ------------------------- tag projection -----------------------------------
// TAG[b][s][t] = h_f . wproj[t,0:256] + h_b . wproj[t,256:512] + bproj[t]
__global__ __launch_bounds__(256)
void proj_kernel(const float* __restrict__ HSEQ,
                 const float* __restrict__ wproj, const float* __restrict__ bproj,
                 float* __restrict__ TAG) {
    __shared__ float wp_s[9 * 512];
    for (int i = threadIdx.x; i < 9 * 512; i += 256) wp_s[i] = wproj[i];
    __syncthreads();

    int row = blockIdx.x * 256 + threadIdx.x;   // 0..32767 = b*512+s
    if (row >= 32768) return;
    const float* hf = HSEQ + (size_t)row * 256;
    const float* hb = HSEQ + ((size_t)64 * 512 + row) * 256;

    float acc[9];
#pragma unroll
    for (int t = 0; t < 9; t++) acc[t] = 0.0f;
    for (int k = 0; k < 256; k += 4) {
        float4 vf = *(const float4*)(hf + k);
        float4 vb = *(const float4*)(hb + k);
#pragma unroll
        for (int t = 0; t < 9; t++) {
            const float* w = wp_s + t * 512;
            acc[t] += vf.x * w[k]       + vf.y * w[k + 1]
                    + vf.z * w[k + 2]   + vf.w * w[k + 3]
                    + vb.x * w[256 + k] + vb.y * w[256 + k + 1]
                    + vb.z * w[256 + k + 2] + vb.w * w[256 + k + 3];
        }
    }
#pragma unroll
    for (int t = 0; t < 9; t++) TAG[(size_t)row * 9 + t] = acc[t] + bproj[t];
}

// ------------------------- CRF log-likelihood -------------------------------
// One wave32 per batch row; lane j<9 carries alpha_j. Mask is all-ones.
__global__ __launch_bounds__(32)
void crf_kernel(const float* __restrict__ TAG, const int* __restrict__ tags,
                const float* __restrict__ startT, const float* __restrict__ endT,
                const float* __restrict__ trans, float* __restrict__ LLH) {
    const int b    = blockIdx.x;
    const int lane = threadIdx.x;
    __shared__ float tr[81];
    for (int i = lane; i < 81; i += 32) tr[i] = trans[i];
    __syncthreads();

    const int jc = (lane < 9) ? lane : 0;
    const float* em = TAG + (size_t)b * 512 * 9;
    const int*   tg = tags + (size_t)b * 512;

    float alpha = startT[jc] + em[jc];
    int   tprev = tg[0];
    float num   = __shfl(alpha, tprev, 32);      // start[tag0] + emit0[tag0]

    for (int t = 1; t < 512; t++) {
        float emt = em[t * 9 + jc];
        float vi[9], m = -1e30f;
#pragma unroll
        for (int i = 0; i < 9; i++) {
            float ai = __shfl(alpha, i, 32);
            vi[i] = ai + tr[i * 9 + jc];
            m = fmaxf(m, vi[i]);
        }
        float ssum = 0.0f;
#pragma unroll
        for (int i = 0; i < 9; i++) ssum += __expf(vi[i] - m);
        alpha = m + __logf(ssum) + emt;

        int   tcur = tg[t];
        float ec   = __shfl(emt, tcur, 32);
        num += tr[tprev * 9 + tcur] + ec;
        tprev = tcur;
    }
    num += endT[tprev];

    float v = (lane < 9) ? (alpha + endT[jc]) : -1e30f;
    float m = v;
    for (int off = 16; off; off >>= 1) m = fmaxf(m, __shfl_xor(m, off, 32));
    float s = __expf(v - m);
    for (int off = 16; off; off >>= 1) s += __shfl_xor(s, off, 32);
    if (lane == 0) LLH[b] = num - (m + __logf(s));
}

__global__ void reduce_kernel(const float* __restrict__ LLH, float* __restrict__ out) {
    if (threadIdx.x == 0) {
        float s = 0.0f;
        for (int i = 0; i < 64; i++) s += LLH[i];
        out[0] = -s / 64.0f;
    }
}

// ------------------------------ launch --------------------------------------
extern "C" void kernel_launch(void* const* d_in, const int* in_sizes, int n_in,
                              void* d_out, int out_size, void* d_ws, size_t ws_size,
                              hipStream_t stream) {
    const float* emb   = (const float*)d_in[0];
    const int*   tags  = (const int*)  d_in[1];
    // d_in[2] attention_masks: all-ones, unused
    const float* wihf  = (const float*)d_in[3];
    const float* whhf  = (const float*)d_in[4];
    const float* bf    = (const float*)d_in[5];
    const float* wihb  = (const float*)d_in[6];
    const float* whhb  = (const float*)d_in[7];
    const float* bb    = (const float*)d_in[8];
    const float* wproj = (const float*)d_in[9];
    const float* bproj = (const float*)d_in[10];
    const float* st    = (const float*)d_in[11];
    const float* et    = (const float*)d_in[12];
    const float* tr    = (const float*)d_in[13];

    char* ws = (char*)d_ws;
    size_t off = 0;
    auto alloc = [&](size_t bytes) {
        size_t o = off;
        off = (off + bytes + 255) & ~(size_t)255;
        return o;
    };
    unsigned short* Wcat = (unsigned short*)(ws + alloc((size_t)2048 * 768 * 2));
    unsigned short* Whf  = (unsigned short*)(ws + alloc((size_t)1024 * 256 * 2));
    unsigned short* Whb  = (unsigned short*)(ws + alloc((size_t)1024 * 256 * 2));
    unsigned short* Xbf  = (unsigned short*)(ws + alloc((size_t)32768 * 768 * 2));
    float*          PRE  = (float*)        (ws + alloc((size_t)32768 * 2048 * 4));
    unsigned short* HBUF = (unsigned short*)(ws + alloc((size_t)2 * 2 * 64 * 256 * 2));
    float*          HSEQ = (float*)        (ws + alloc((size_t)2 * 64 * 512 * 256 * 4));
    float*          TAG  = (float*)        (ws + alloc((size_t)64 * 512 * 9 * 4));
    float*          LLH  = (float*)        (ws + alloc((size_t)64 * 4));
    unsigned int*   BAR  = (unsigned int*) (ws + alloc(64));

    init_kernel<<<1, 32, 0, stream>>>(BAR);
    convert_weights<<<8192, 256, 0, stream>>>(wihf, wihb, whhf, whhb, Wcat, Whf, Whb);
    convert_x<<<12288, 256, 0, stream>>>(emb, Xbf);
    gemm_pre<<<512 * 32, 128, 0, stream>>>(Xbf, Wcat, bf, bb, PRE);
    lstm_kernel<<<16, 256, 0, stream>>>(PRE, Whf, Whb, HBUF, HSEQ, BAR);
    proj_kernel<<<128, 256, 0, stream>>>(HSEQ, wproj, bproj, TAG);
    crf_kernel<<<64, 32, 0, stream>>>(TAG, tags, st, et, tr, LLH);
    reduce_kernel<<<1, 32, 0, stream>>>(LLH, (float*)d_out);
}
